// ActorTransformer_81862076662200
// MI455X (gfx1250) — compile-verified, matrix operand order checked by reading, not run on hardware
//
#include <hip/hip_runtime.h>

// ---------------------------------------------------------------------------
// ActorTransformer forward for gfx1250 (MI455X, wave32, WMMA).
// - All GEMMs on v_wmma_f32_16x16x32_f16 (f16 in, f32 accumulate),
//   16x64 output tile per wave, software-pipelined K loop.
// - Attention: one wave per (atom, head); Q/K/V staged to LDS via the
//   Tensor Data Mover (tensor_load_to_lds + s_wait_tensorcnt) when available.
// Scratch arena requirement: ~118 MB in d_ws.
// ---------------------------------------------------------------------------

#define BGR      64      // graphs (B)
#define ATM      512     // atoms per graph (A)
#define DIMN     256     // model dim
#define MTOK     (BGR*ATM)
#define FFN      2048
#define NHEADS   8
#define DH       32
#define NLAYERS  6
#define TOR      2016
#define NACT     6
#define S2STEPS  6

typedef __attribute__((ext_vector_type(16))) _Float16      v16h;
typedef __attribute__((ext_vector_type(8)))  float         v8f;
typedef __attribute__((ext_vector_type(8)))  unsigned int  v8u;
typedef __attribute__((ext_vector_type(4)))  unsigned int  v4u;
typedef __attribute__((ext_vector_type(8)))  int           v8i;
typedef __attribute__((ext_vector_type(4)))  int           v4i;

#if defined(__has_builtin)
#  if __has_builtin(__builtin_amdgcn_tensor_load_to_lds)
#    define HAVE_TDM 1
#  endif
#endif
#ifndef HAVE_TDM
#  define HAVE_TDM 0
#endif

// K index (even) of VGPR pair v for this lane-half, per CDNA5 16-bit A/B layout
__device__ __forceinline__ int kpair_of(int v, int half) {
  return (v < 4) ? (v * 2 + half * 8) : (16 + (v - 4) * 2 + half * 8);
}

// Load a 16x32 f16 fragment where element [r][k] = base[(row+r)*ld + k0 + k].
// Works for A-matrix rows and for B == W stored as (N x K) row-major (x @ W^T).
__device__ __forceinline__ v16h load_frag(const _Float16* base, int row, int ld,
                                          int k0, int lane) {
  int r = lane & 15;
  int half = (lane >> 4) & 1;
  const _Float16* p = base + (size_t)(row + r) * ld + k0;
  v8u w;
#pragma unroll
  for (int v = 0; v < 8; ++v) {
    int k = kpair_of(v, half);
    w[v] = *reinterpret_cast<const unsigned int*>(p + k);  // even k -> dword aligned
  }
  return __builtin_bit_cast(v16h, w);
}

// Load a B fragment where logical B[n][k] = base[(k0+k)*ld + col0 + n]  (K-major source)
__device__ __forceinline__ v16h load_frag_t(const _Float16* base, int col0, int ld,
                                            int k0, int lane) {
  int nloc = lane & 15;
  int half = (lane >> 4) & 1;
  v16h out;
#pragma unroll
  for (int v = 0; v < 8; ++v) {
    int k = kpair_of(v, half);
    out[2 * v]     = base[(size_t)(k0 + k) * ld + col0 + nloc];
    out[2 * v + 1] = base[(size_t)(k0 + k + 1) * ld + col0 + nloc];
  }
  return out;
}

#if HAVE_TDM
// Issue a TDM 2D-tile load: tile_h rows x tile_w f16 elements, global row
// stride `row_stride` (in elements), packed contiguously at LDS offset lds_off.
// D# layout per CDNA5 ISA 8.3/8.4 (group0: 128b, group1: 256b; groups 2/3 zero).
__device__ __forceinline__ void tdm_load_tile_2d(unsigned lds_off,
                                                 const _Float16* gptr,
                                                 unsigned tile_w, unsigned tile_h,
                                                 unsigned long long row_stride) {
  unsigned long long ga = (unsigned long long)(void*)gptr;
  v4u g0;
  g0[0] = 1u;                                              // count=1, user mode
  g0[1] = lds_off;                                         // lds_addr (bytes)
  g0[2] = (unsigned)(ga & 0xffffffffu);                    // global_addr[31:0]
  g0[3] = (unsigned)((ga >> 32) & 0x01ffffffu) | (2u << 30); // ga[56:32] | type=2
  v8i g1;
  g1[0] = 1 << 16;                                         // data_size = 1 (2 bytes)
  g1[1] = (int)((tile_w & 0xffffu) << 16);                 // tensor_dim0 lo
  g1[2] = (int)((tile_w >> 16) | ((tile_h & 0xffffu) << 16)); // td0 hi | td1 lo
  g1[3] = (int)((tile_h >> 16) | ((tile_w & 0xffffu) << 16)); // td1 hi | tile_dim0
  g1[4] = (int)(tile_h & 0xffffu);                         // tile_dim1 (tile_dim2=0)
  g1[5] = (int)(row_stride & 0xffffffffu);                 // tensor_dim0_stride lo
  g1[6] = (int)((row_stride >> 32) & 0xffffu);             // stride hi (dim1_stride=0)
  g1[7] = 0;
  v4i z4 = {0, 0, 0, 0};
#if __clang_major__ >= 23
  v8i z8 = {0, 0, 0, 0, 0, 0, 0, 0};
  __builtin_amdgcn_tensor_load_to_lds(g0, g1, z4, z4, z8, 0);
#else
  __builtin_amdgcn_tensor_load_to_lds(g0, g1, z4, z4, 0);
#endif
}
#endif  // HAVE_TDM

// ----------------------------- elementwise utils ---------------------------

__global__ void k_f32_to_f16(const float* __restrict__ src, _Float16* __restrict__ dst, int n) {
  int i = blockIdx.x * blockDim.x + threadIdx.x;
  if (i < n) dst[i] = (_Float16)src[i];
}

__global__ void k_zero_f32(float* __restrict__ p, int n) {
  int i = blockIdx.x * blockDim.x + threadIdx.x;
  if (i < n) p[i] = 0.f;
}

__global__ void k_copy_f32(float* __restrict__ dst, const float* __restrict__ src, int n) {
  int i = blockIdx.x * blockDim.x + threadIdx.x;
  if (i < n) dst[i] = src[i];
}

// ----------------------------- lin0 + relu ---------------------------------
__global__ void k_lin0(const float* __restrict__ data, const float* __restrict__ w,
                       const float* __restrict__ b, _Float16* __restrict__ xh) {
  int m = blockIdx.x, d = threadIdx.x;
  float acc = b[d];
#pragma unroll
  for (int j = 0; j < 3; ++j) acc += data[(size_t)m * 3 + j] * w[d * 3 + j];
  acc = fmaxf(acc, 0.f);
  xh[(size_t)m * DIMN + d] = (_Float16)acc;
}

// ----------------------------- WMMA GEMM -----------------------------------
// C(MxN) = A(MxK,f16) @ W(NxK,f16)^T + bias, optional relu; f32 and/or f16 out.
// One wave computes a 16x64 tile (4 accumulators reuse the A fragment);
// K loop is software-pipelined (next fragments loaded before current WMMAs).
__global__ void k_gemm(const _Float16* __restrict__ A, const _Float16* __restrict__ W,
                       const float* __restrict__ bias, float* __restrict__ Cf,
                       _Float16* __restrict__ Ch, int M, int N, int K, int relu) {
  int wid = blockIdx.x * (blockDim.x >> 5) + (threadIdx.x >> 5);
  int ngroups = N >> 6;                       // 64-wide column groups
  int total = (M >> 4) * ngroups;
  if (wid >= total) return;
  int mt = wid / ngroups, ng = wid % ngroups;
  int lane = threadIdx.x & 31;
  int nn = lane & 15;
  int half = lane >> 4;
  const _Float16* Ab = A + (size_t)(mt * 16) * K;
  const _Float16* Wb = W + ((size_t)ng * 64) * K;

  v8f c[4];
#pragma unroll
  for (int j = 0; j < 4; ++j) {
    float bv = bias ? bias[ng * 64 + j * 16 + nn] : 0.f;
    v8f t = {bv, bv, bv, bv, bv, bv, bv, bv};
    c[j] = t;
  }

  v16h a0 = load_frag(Ab, 0, K, 0, lane);
  v16h b0[4];
#pragma unroll
  for (int j = 0; j < 4; ++j) b0[j] = load_frag(Wb, j * 16, K, 0, lane);

  for (int k0 = 32; k0 < K; k0 += 32) {
    __builtin_prefetch(Ab + (size_t)nn * K + k0 + 32, 0, 1);  // global_prefetch_b8
    v16h a1 = load_frag(Ab, 0, K, k0, lane);
    v16h b1[4];
#pragma unroll
    for (int j = 0; j < 4; ++j) b1[j] = load_frag(Wb, j * 16, K, k0, lane);
#pragma unroll
    for (int j = 0; j < 4; ++j)
      c[j] = __builtin_amdgcn_wmma_f32_16x16x32_f16(false, a0, false, b0[j], (short)0,
                                                    c[j], false, false);
    a0 = a1;
#pragma unroll
    for (int j = 0; j < 4; ++j) b0[j] = b1[j];
  }
#pragma unroll
  for (int j = 0; j < 4; ++j)
    c[j] = __builtin_amdgcn_wmma_f32_16x16x32_f16(false, a0, false, b0[j], (short)0,
                                                  c[j], false, false);

#pragma unroll
  for (int j = 0; j < 4; ++j) {
#pragma unroll
    for (int v = 0; v < 8; ++v) {
      int m = mt * 16 + v + 8 * half;
      int nq = ng * 64 + j * 16 + nn;
      float val = c[j][v];
      if (relu) val = fmaxf(val, 0.f);
      if (Cf) Cf[(size_t)m * N + nq] = val;
      if (Ch) Ch[(size_t)m * N + nq] = (_Float16)val;
    }
  }
}

// ----------------------------- attention -----------------------------------
// One wave per (atom n, head h). Sequence axis = graphs (S = 64), dh = 32.
__global__ void __launch_bounds__(32) k_attn(const _Float16* __restrict__ qkv,
                                             _Float16* __restrict__ ah) {
  int n = blockIdx.x >> 3;
  int h = blockIdx.x & 7;
  int lane = threadIdx.x;
  __shared__ _Float16 Qs[64 * 32];
  __shared__ _Float16 Ks[64 * 32];
  __shared__ _Float16 Vs[64 * 32];
  __shared__ float    Sf[64 * 64];
  __shared__ _Float16 Ph[64 * 64];

  const _Float16* gq = qkv + (size_t)n * (3 * DIMN) + h * DH;  // row s=0
#if HAVE_TDM
  // TDM: 64 rows (graphs) x 32 f16 each, row stride = 3*DIMN*ATM elements.
  const unsigned long long rs = (unsigned long long)(3 * DIMN) * ATM;
  tdm_load_tile_2d((unsigned)(unsigned long long)(void*)Qs, gq,            32, 64, rs);
  tdm_load_tile_2d((unsigned)(unsigned long long)(void*)Ks, gq + DIMN,     32, 64, rs);
  tdm_load_tile_2d((unsigned)(unsigned long long)(void*)Vs, gq + 2 * DIMN, 32, 64, rs);
#  if __has_builtin(__builtin_amdgcn_s_wait_tensorcnt)
  __builtin_amdgcn_s_wait_tensorcnt(0);
#  endif
  __syncthreads();
#else
  for (int idx = lane; idx < 64 * 32; idx += 32) {
    int s = idx >> 5, d = idx & 31;
    size_t base = ((size_t)s * ATM) * (3 * DIMN) + d;
    Qs[idx] = gq[base];
    Ks[idx] = gq[base + DIMN];
    Vs[idx] = gq[base + 2 * DIMN];
  }
  __syncthreads();
#endif

  const float scale = 0.17677669529663687f;  // 1/sqrt(32)
  int nn = lane & 15, half = lane >> 4;

  // S = Q @ K^T (four 16-row x four 16-col tiles, single K-step of 32)
  for (int st = 0; st < 4; ++st)
    for (int tt = 0; tt < 4; ++tt) {
      v16h a = load_frag(Qs, st * 16, 32, 0, lane);
      v16h b = load_frag(Ks, tt * 16, 32, 0, lane);
      v8f c = {};
      c = __builtin_amdgcn_wmma_f32_16x16x32_f16(false, a, false, b, (short)0, c,
                                                 false, false);
#pragma unroll
      for (int v = 0; v < 8; ++v) {
        int s = st * 16 + v + 8 * half;
        int t = tt * 16 + nn;
        Sf[s * 64 + t] = c[v] * scale;
      }
    }
  __syncthreads();

  // row softmax over t
  for (int r = lane; r < 64; r += 32) {
    float mx = -1e30f;
    for (int t = 0; t < 64; ++t) mx = fmaxf(mx, Sf[r * 64 + t]);
    float sum = 0.f;
    for (int t = 0; t < 64; ++t) {
      float e = __expf(Sf[r * 64 + t] - mx);
      sum += e;
      Sf[r * 64 + t] = e;
    }
    float inv = 1.f / sum;
    for (int t = 0; t < 64; ++t) Ph[r * 64 + t] = (_Float16)(Sf[r * 64 + t] * inv);
  }
  __syncthreads();

  // O = P @ V  (P: 64x64, V: 64x32 -> B needs V^T indexing)
  for (int st = 0; st < 4; ++st)
    for (int dt = 0; dt < 2; ++dt) {
      v8f c = {};
      for (int kk = 0; kk < 64; kk += 32) {
        v16h a = load_frag(Ph, st * 16, 64, kk, lane);
        v16h b = load_frag_t(Vs, dt * 16, 32, kk, lane);
        c = __builtin_amdgcn_wmma_f32_16x16x32_f16(false, a, false, b, (short)0, c,
                                                   false, false);
      }
#pragma unroll
      for (int v = 0; v < 8; ++v) {
        int s = st * 16 + v + 8 * half;
        int d = dt * 16 + nn;
        ah[((size_t)s * ATM + n) * DIMN + h * DH + d] = (_Float16)c[v];
      }
    }
}

// ----------------------------- residual + LayerNorm ------------------------
__global__ void k_add_ln(_Float16* __restrict__ xh, const _Float16* __restrict__ yh,
                         const float* __restrict__ g, const float* __restrict__ bb) {
  int m = blockIdx.x, d = threadIdx.x;
  __shared__ float red[256];
  size_t i = (size_t)m * DIMN + d;
  float t = (float)xh[i] + (float)yh[i];
  red[d] = t;
  __syncthreads();
  for (int s = 128; s > 0; s >>= 1) {
    if (d < s) red[d] += red[d + s];
    __syncthreads();
  }
  float mean = red[0] * (1.f / DIMN);
  __syncthreads();
  float dv = t - mean;
  red[d] = dv * dv;
  __syncthreads();
  for (int s = 128; s > 0; s >>= 1) {
    if (d < s) red[d] += red[d + s];
    __syncthreads();
  }
  float var = red[0] * (1.f / DIMN);
  float o = dv * rsqrtf(var + 1e-5f) * g[d] + bb[d];
  xh[i] = (_Float16)o;
}

// ----------------------------- LSTM step -----------------------------------
// torch gate order i,f,g,o; in_dim fixed at 2*DIMN (q_star).
__global__ void k_lstm(const float* __restrict__ q, const float* __restrict__ hin,
                       const float* __restrict__ cin, const float* __restrict__ wih,
                       const float* __restrict__ whh, const float* __restrict__ bih,
                       const float* __restrict__ bhh, float* __restrict__ hout,
                       float* __restrict__ cout) {
  int b = blockIdx.x, d = threadIdx.x;
  __shared__ float hs[DIMN];
  __shared__ float qs[2 * DIMN];
  hs[d] = hin[(size_t)b * DIMN + d];
  qs[d] = q[(size_t)b * 2 * DIMN + d];
  qs[d + DIMN] = q[(size_t)b * 2 * DIMN + d + DIMN];
  __syncthreads();
  float gate[4];
#pragma unroll
  for (int gi = 0; gi < 4; ++gi) {
    int j = gi * DIMN + d;
    float acc = bih[j] + bhh[j];
    const float* wi = wih + (size_t)j * 2 * DIMN;
    const float* wh = whh + (size_t)j * DIMN;
    for (int k = 0; k < 2 * DIMN; ++k) acc += qs[k] * wi[k];
    for (int k = 0; k < DIMN; ++k) acc += hs[k] * wh[k];
    gate[gi] = acc;
  }
  float ig = 1.f / (1.f + __expf(-gate[0]));
  float fg = 1.f / (1.f + __expf(-gate[1]));
  float gg = tanhf(gate[2]);
  float og = 1.f / (1.f + __expf(-gate[3]));
  float cnew = fg * cin[(size_t)b * DIMN + d] + ig * gg;
  hout[(size_t)b * DIMN + d] = og * tanhf(cnew);
  cout[(size_t)b * DIMN + d] = cnew;
}

// ----------------------------- Set2Set attention step ----------------------
__global__ void k_s2s(const _Float16* __restrict__ x, const float* __restrict__ h,
                      float* __restrict__ qstar) {
  int b = blockIdx.x, tid = threadIdx.x;
  __shared__ float hs[DIMN];
  __shared__ float es[ATM];
  __shared__ float red[256];
  hs[tid] = h[(size_t)b * DIMN + tid];
  __syncthreads();
  for (int a = tid; a < ATM; a += 256) {
    const _Float16* xr = x + ((size_t)b * ATM + a) * DIMN;
    float acc = 0.f;
    for (int d = 0; d < DIMN; ++d) acc += (float)xr[d] * hs[d];
    es[a] = acc;
  }
  __syncthreads();
  red[tid] = fmaxf(es[tid], es[tid + 256]);
  __syncthreads();
  for (int s = 128; s > 0; s >>= 1) {
    if (tid < s) red[tid] = fmaxf(red[tid], red[tid + s]);
    __syncthreads();
  }
  float mx = red[0];
  __syncthreads();
  float e0 = __expf(es[tid] - mx), e1 = __expf(es[tid + 256] - mx);
  es[tid] = e0;
  es[tid + 256] = e1;
  red[tid] = e0 + e1;
  __syncthreads();
  for (int s = 128; s > 0; s >>= 1) {
    if (tid < s) red[tid] += red[tid + s];
    __syncthreads();
  }
  float inv = 1.f / red[0];
  __syncthreads();
  float r = 0.f;
  for (int a = 0; a < ATM; ++a)
    r += es[a] * (float)x[((size_t)b * ATM + a) * DIMN + tid];
  r *= inv;
  qstar[(size_t)b * 2 * DIMN + tid] = hs[tid];
  qstar[(size_t)b * 2 * DIMN + DIMN + tid] = r;
}

// ----------------------------- feature build -------------------------------
__global__ void k_feat(const _Float16* __restrict__ x, const float* __restrict__ h1,
                       const int* __restrict__ nonring, const int* __restrict__ nrbidx,
                       _Float16* __restrict__ feath) {
  int i = blockIdx.x * blockDim.x + threadIdx.x;
  if (i >= TOR * 5 * DIMN) return;
  int d = i / (TOR * 5);
  int rem = i % (TOR * 5);
  int t = rem / 5;
  int s = rem % 5;
  float val;
  if (s == 0) {
    val = h1[(size_t)nrbidx[t] * DIMN + d];
  } else {
    int idx = nonring[t * 4 + (s - 1)];
    val = (float)x[(size_t)idx * DIMN + d];
  }
  feath[i] = (_Float16)val;
}

// ----------------------------- lin2 head (N=6, scalar) ---------------------
__global__ void k_lin2(const _Float16* __restrict__ r1, const float* __restrict__ w,
                       const float* __restrict__ bias, float* __restrict__ o) {
  int i = blockIdx.x * blockDim.x + threadIdx.x;
  if (i >= TOR * NACT) return;
  int t = i / NACT, j = i % NACT;
  float acc = bias[j];
  const _Float16* rr = r1 + (size_t)t * DIMN;
  const float* wr = w + (size_t)j * DIMN;
  for (int d = 0; d < DIMN; ++d) acc += (float)rr[d] * wr[d];
  o[i] = acc;
}

// ----------------------------- scatter to padded output --------------------
__global__ void k_offsets(const int* __restrict__ sizes, int* __restrict__ off) {
  if (blockIdx.x == 0 && threadIdx.x == 0) {
    int acc = 0;
    for (int b = 0; b < BGR; ++b) { off[b] = acc; acc += sizes[b]; }
    off[BGR] = acc;
  }
}

__global__ void k_scatter(const float* __restrict__ o, const int* __restrict__ off,
                          float* __restrict__ out, int max_s) {
  int t = blockIdx.x * blockDim.x + threadIdx.x;
  if (t >= TOR) return;
  int b = 0;
  while (b < BGR - 1 && off[b + 1] <= t) ++b;
  int pos = t - off[b];
  for (int j = 0; j < NACT; ++j)
    out[((size_t)b * max_s + pos) * NACT + j] = o[t * NACT + j];
}

// ===========================================================================

extern "C" void kernel_launch(void* const* d_in, const int* in_sizes, int n_in,
                              void* d_out, int out_size, void* d_ws, size_t ws_size,
                              hipStream_t stream) {
  (void)in_sizes; (void)n_in; (void)ws_size;

  const float* data      = (const float*)d_in[0];
  const int*   nonring   = (const int*)d_in[1];
  const int*   nrbidx    = (const int*)d_in[2];
  const int*   sizes     = (const int*)d_in[3];
  const float* lin0_w    = (const float*)d_in[4];
  const float* lin0_b    = (const float*)d_in[5];
  const float* qkv_w     = (const float*)d_in[6];
  const float* qkv_b     = (const float*)d_in[7];
  const float* out_w     = (const float*)d_in[8];
  const float* out_b     = (const float*)d_in[9];
  const float* ff1_w     = (const float*)d_in[10];
  const float* ff1_b     = (const float*)d_in[11];
  const float* ff2_w     = (const float*)d_in[12];
  const float* ff2_b     = (const float*)d_in[13];
  const float* ln1_g     = (const float*)d_in[14];
  const float* ln1_b     = (const float*)d_in[15];
  const float* ln2_g     = (const float*)d_in[16];
  const float* ln2_b     = (const float*)d_in[17];
  const float* s2s_wih   = (const float*)d_in[18];
  const float* s2s_whh   = (const float*)d_in[19];
  const float* s2s_bih   = (const float*)d_in[20];
  const float* s2s_bhh   = (const float*)d_in[21];
  const float* mem_wih   = (const float*)d_in[22];
  const float* mem_whh   = (const float*)d_in[23];
  const float* mem_bih   = (const float*)d_in[24];
  const float* mem_bhh   = (const float*)d_in[25];
  const float* lin1_w    = (const float*)d_in[26];
  const float* lin1_b    = (const float*)d_in[27];
  const float* lin2_w    = (const float*)d_in[28];
  const float* lin2_b    = (const float*)d_in[29];

  float* out = (float*)d_out;
  int max_s = (out_size - 2 * BGR * DIMN) / (BGR * NACT);  // = 63
  int logitN = BGR * max_s * NACT;

  // ---- scratch arena (~118 MB) ----
  char* wsp = (char*)d_ws;
  size_t off = 0;
  auto alloc = [&](size_t bytes) -> void* {
    void* p = wsp + off;
    off = (off + bytes + 255) & ~(size_t)255;
    return p;
  };
  _Float16* xh     = (_Float16*)alloc((size_t)MTOK * DIMN * 2);
  _Float16* yh     = (_Float16*)alloc((size_t)MTOK * DIMN * 2);
  _Float16* qkvh   = (_Float16*)alloc((size_t)MTOK * 3 * DIMN * 2);
  _Float16* ahbuf  = (_Float16*)alloc((size_t)MTOK * DIMN * 2);
  _Float16* ffh    = (_Float16*)alloc((size_t)2048 * FFN * 2);
  _Float16* wh1    = (_Float16*)alloc((size_t)FFN * DIMN * 2);
  _Float16* wh2    = (_Float16*)alloc((size_t)FFN * DIMN * 2);
  _Float16* feath  = (_Float16*)alloc((size_t)TOR * 5 * DIMN * 2);
  _Float16* relu1h = (_Float16*)alloc((size_t)TOR * DIMN * 2);
  float*    qstar  = (float*)alloc((size_t)BGR * 2 * DIMN * 4);
  float*    hbuf   = (float*)alloc((size_t)BGR * DIMN * 4);
  float*    cbuf   = (float*)alloc((size_t)BGR * DIMN * 4);
  float*    h1     = (float*)alloc((size_t)BGR * DIMN * 4);
  float*    c1     = (float*)alloc((size_t)BGR * DIMN * 4);
  float*    obuf   = (float*)alloc((size_t)TOR * NACT * 4);
  int*      offs   = (int*)alloc((BGR + 1) * 4);

  auto cvt = [&](const float* s, _Float16* d, int n) {
    k_f32_to_f16<<<(n + 255) / 256, 256, 0, stream>>>(s, d, n);
  };
  auto gemm = [&](const _Float16* A, const _Float16* W, const float* bias,
                  float* Cf, _Float16* Ch, int M, int N, int K, int relu) {
    int waves = (M / 16) * (N / 64);          // one 16x64 tile per wave
    int blocks = (waves + 7) / 8;             // 8 waves (256 threads) per block
    k_gemm<<<blocks, 256, 0, stream>>>(A, W, bias, Cf, Ch, M, N, K, relu);
  };
  auto zero = [&](float* p, int n) {
    k_zero_f32<<<(n + 255) / 256, 256, 0, stream>>>(p, n);
  };

  // ---- lin0 ----
  k_lin0<<<MTOK, 256, 0, stream>>>(data, lin0_w, lin0_b, xh);

  // ---- transformer encoder ----
  for (int l = 0; l < NLAYERS; ++l) {
    cvt(qkv_w + (size_t)l * 3 * DIMN * DIMN, wh1, 3 * DIMN * DIMN);
    gemm(xh, wh1, qkv_b + l * 3 * DIMN, nullptr, qkvh, MTOK, 3 * DIMN, DIMN, 0);
    k_attn<<<ATM * NHEADS, 32, 0, stream>>>(qkvh, ahbuf);
    cvt(out_w + (size_t)l * DIMN * DIMN, wh1, DIMN * DIMN);
    gemm(ahbuf, wh1, out_b + l * DIMN, nullptr, yh, MTOK, DIMN, DIMN, 0);
    k_add_ln<<<MTOK, 256, 0, stream>>>(xh, yh, ln1_g + l * DIMN, ln1_b + l * DIMN);
    cvt(ff1_w + (size_t)l * FFN * DIMN, wh1, FFN * DIMN);
    cvt(ff2_w + (size_t)l * DIMN * FFN, wh2, DIMN * FFN);
    for (int c = 0; c < MTOK / 2048; ++c) {
      _Float16* xc = xh + (size_t)c * 2048 * DIMN;
      _Float16* yc = yh + (size_t)c * 2048 * DIMN;
      gemm(xc, wh1, ff1_b + l * FFN, nullptr, ffh, 2048, FFN, DIMN, 1);
      gemm(ffh, wh2, ff2_b + l * DIMN, nullptr, yc, 2048, DIMN, FFN, 0);
    }
    k_add_ln<<<MTOK, 256, 0, stream>>>(xh, yh, ln2_g + l * DIMN, ln2_b + l * DIMN);
  }

  // ---- Set2Set pooling ----
  zero(qstar, BGR * 2 * DIMN);
  zero(hbuf, BGR * DIMN);
  zero(cbuf, BGR * DIMN);
  for (int s = 0; s < S2STEPS; ++s) {
    k_lstm<<<BGR, 256, 0, stream>>>(qstar, hbuf, cbuf, s2s_wih, s2s_whh, s2s_bih,
                                    s2s_bhh, hbuf, cbuf);
    k_s2s<<<BGR, 256, 0, stream>>>(xh, hbuf, qstar);
  }

  // ---- memory LSTM (zero init) ----
  zero(hbuf, BGR * DIMN);
  zero(cbuf, BGR * DIMN);
  k_lstm<<<BGR, 256, 0, stream>>>(qstar, hbuf, cbuf, mem_wih, mem_whh, mem_bih,
                                  mem_bhh, h1, c1);
  k_copy_f32<<<(BGR * DIMN + 255) / 256, 256, 0, stream>>>(out + logitN, h1, BGR * DIMN);
  k_copy_f32<<<(BGR * DIMN + 255) / 256, 256, 0, stream>>>(out + logitN + BGR * DIMN,
                                                           c1, BGR * DIMN);

  // ---- feature build + head ----
  k_feat<<<(TOR * 5 * DIMN + 255) / 256, 256, 0, stream>>>(xh, h1, nonring, nrbidx,
                                                           feath);
  cvt(lin1_w, wh1, DIMN * 5 * DIMN);
  gemm(feath, wh1, lin1_b, nullptr, relu1h, TOR, DIMN, 5 * DIMN, 1);
  k_lin2<<<(TOR * NACT + 255) / 256, 256, 0, stream>>>(relu1h, lin2_w, lin2_b, obuf);

  // ---- scatter into padded logits ----
  k_offsets<<<1, 1, 0, stream>>>(sizes, offs);
  k_zero_f32<<<(logitN + 255) / 256, 256, 0, stream>>>(out, logitN);
  k_scatter<<<(TOR + 255) / 256, 256, 0, stream>>>(obuf, offs, out, max_s);
}
